// SANet_51694226374934
// MI455X (gfx1250) — matrix operand order repaired and checked by reading, not compiled, not run on hardware
//
#include <hip/hip_runtime.h>

// ---------------------------------------------------------------------------
// SANet attention block for MI455X (gfx1250, wave32, WMMA).
// Instance-norm is folded into per-batch adjusted conv weights, so the whole
// network is 6 bf16-WMMA GEMMs + one softmax pass. fp32 accumulate throughout.
// ---------------------------------------------------------------------------

typedef __attribute__((ext_vector_type(16))) __bf16 v16bf;
typedef __attribute__((ext_vector_type(8)))  float  v8f;

#define LDSK 48          // padded LDS row stride (bf16 units): 96B, 16B-aligned

union Frag { v16bf v; uint4 q[2]; };
union Pk4  { __bf16 h[4]; unsigned long long u; };

// ---------------- per-(b,c) row statistics:  alpha = rsqrt(var+eps), beta = -mean*alpha
__global__ __launch_bounds__(256) void rowstats_kernel(const float* __restrict__ x,
                                                       float* __restrict__ alpha,
                                                       float* __restrict__ beta, int T) {
  const long row = blockIdx.x;
  const float* xr = x + row * (long)T;
  float s = 0.f, s2 = 0.f;
  for (int i = threadIdx.x; i < T; i += 256) { float v = xr[i]; s += v; s2 += v * v; }
  __shared__ float sh[512];
  sh[threadIdx.x] = s; sh[256 + threadIdx.x] = s2; __syncthreads();
  for (int o = 128; o > 0; o >>= 1) {
    if (threadIdx.x < (unsigned)o) {
      sh[threadIdx.x] += sh[threadIdx.x + o];
      sh[256 + threadIdx.x] += sh[256 + threadIdx.x + o];
    }
    __syncthreads();
  }
  if (threadIdx.x == 0) {
    float m = sh[0] / (float)T;
    float v = sh[256] / (float)T - m * m;           // biased variance
    float a = rsqrtf(v + 1e-5f);
    alpha[row] = a; beta[row] = -m * a;
  }
}

// ---------------- fold inorm into weights: W'[b,o,c] = W[o,c]*alpha[b,c],
//                  b'[b,o] = b[o] + sum_c W[o,c]*beta[b,c]
__global__ __launch_bounds__(256) void wadjust_kernel(const float* __restrict__ W,
                                                      const float* __restrict__ bias,
                                                      const float* __restrict__ alpha,
                                                      const float* __restrict__ beta,
                                                      float* __restrict__ Wadj,
                                                      float* __restrict__ badj, int C) {
  const int o = blockIdx.x, b = blockIdx.y;
  const float* al = alpha + (long)b * C;
  const float* be = beta  + (long)b * C;
  float acc = 0.f;
  for (int c = threadIdx.x; c < C; c += 256) {
    float w = W[(long)o * C + c];
    Wadj[((long)b * C + o) * C + c] = w * al[c];
    acc += w * be[c];
  }
  __shared__ float sh[256];
  sh[threadIdx.x] = acc; __syncthreads();
  for (int off = 128; off > 0; off >>= 1) {
    if (threadIdx.x < (unsigned)off) sh[threadIdx.x] += sh[threadIdx.x + off];
    __syncthreads();
  }
  if (threadIdx.x == 0) badj[(long)b * C + o] = bias[o] + sh[0];
}

// ---------------- WMMA GEMM: D[b] = A[b] (MxK) * B[b] (KxN) (+bias[b])(+resid[b])
// TRANSA: A stored [K][M] (lda = M-row stride). TRANSB: B stored [N][K] (ldb = K-row stride).
// 128x128 tile / WG, 8 waves, each wave 32(M)x64(N) via 2x4 grid of 16x16x32 bf16 WMMAs.
template <bool TRANSA, bool TRANSB, bool BIAS, bool RESID>
__global__ __launch_bounds__(256) void gemm_wmma_kernel(
    const float* __restrict__ Ag, long aBatch,
    const float* __restrict__ Bg, long bBatch,
    float* __restrict__ Dg, long dBatch,
    const float* __restrict__ biasg, long biasBatch,
    const float* __restrict__ residg, long rBatch,
    int M, int N, int K, int lda, int ldb, int ldd) {
  __shared__ __bf16 As[128][LDSK];   // logical [m][k], k < 32
  __shared__ __bf16 Bs[128][LDSK];   // logical [n][k], k < 32

  const int b = blockIdx.z;
  const float* A  = Ag + (long)b * aBatch;
  const float* Bm = Bg + (long)b * bBatch;
  float*       D  = Dg + (long)b * dBatch;
  const int m0 = blockIdx.y * 128, n0 = blockIdx.x * 128;
  const int tid  = threadIdx.x;
  const int lane = tid & 31, wave = tid >> 5;
  const int wm = wave & 3;      // 4 M-subtiles of 32 rows
  const int wn = wave >> 2;     // 2 N-subtiles of 64 cols
  const int half = lane >> 4;   // lane half-select (CDNA5 16-bit frag layout)
  const int l16  = lane & 15;

  v8f zero = {};
  v8f acc[2][4];
  for (int i = 0; i < 2; i++) for (int j = 0; j < 4; j++) acc[i][j] = zero;

  for (int k0 = 0; k0 < K; k0 += 32) {
    __syncthreads();   // protect LDS from previous iteration's readers
    // ---- stage A tile (fp32 -> bf16) into As[m][k]
    if (!TRANSA) {
      const int kk = (tid & 7) * 4, mb = tid >> 3;     // 4 consecutive k per thread
      for (int p = 0; p < 4; p++) {
        int m = mb + p * 32;
        float4 f = *(const float4*)(A + (long)(m0 + m) * lda + k0 + kk);
        Pk4 t; t.h[0] = (__bf16)f.x; t.h[1] = (__bf16)f.y;
               t.h[2] = (__bf16)f.z; t.h[3] = (__bf16)f.w;
        *(unsigned long long*)&As[m][kk] = t.u;
      }
    } else {                                            // A is [K][M]: coalesce over m
      const int mm = tid & 127, k1 = tid >> 7;
      for (int k = k1; k < 32; k += 2)
        As[mm][k] = (__bf16)A[(long)(k0 + k) * lda + m0 + mm];
    }
    // ---- stage B tile into Bs[n][k] (transposed so frags are contiguous)
    if (!TRANSB) {                                      // B is [K][N]: coalesce over n
      const int nn = tid & 127, k1 = tid >> 7;
      for (int k = k1; k < 32; k += 2)
        Bs[nn][k] = (__bf16)Bm[(long)(k0 + k) * ldb + n0 + nn];
    } else {                                            // B is [N][K]: rows already n-major
      const int n = tid >> 1, kk = (tid & 1) * 16;
      for (int q = 0; q < 4; q++) {
        float4 f = *(const float4*)(Bm + (long)(n0 + n) * ldb + k0 + kk + q * 4);
        Pk4 t; t.h[0] = (__bf16)f.x; t.h[1] = (__bf16)f.y;
               t.h[2] = (__bf16)f.z; t.h[3] = (__bf16)f.w;
        *(unsigned long long*)&Bs[n][kk + q * 4] = t.u;
      }
    }
    __syncthreads();

    // ---- build fragments per CDNA5 ISA layouts (CDNA5 ISA §7.12.2) and multiply
    Frag afr[2], bfr[4];
    for (int i = 0; i < 2; i++) {
      const __bf16* rp = &As[wm * 32 + i * 16 + l16][0];
      afr[i].q[0] = *(const uint4*)(rp + half * 8);       // K 0..7  / 8..15
      afr[i].q[1] = *(const uint4*)(rp + 16 + half * 8);  // K 16..23 / 24..31
    }
    for (int j = 0; j < 4; j++) {
      const __bf16* rp = &Bs[wn * 64 + j * 16 + l16][0];
      bfr[j].q[0] = *(const uint4*)(rp + half * 8);       // K 0..7  / 8..15
      bfr[j].q[1] = *(const uint4*)(rp + 16 + half * 8);  // K 16..23 / 24..31
    }
    for (int i = 0; i < 2; i++)
      for (int j = 0; j < 4; j++)
        acc[i][j] = __builtin_amdgcn_wmma_f32_16x16x32_bf16(
            false, afr[i].v, false, bfr[j].v, (short)0, acc[i][j], false, false);
  }

  // ---- epilogue: D layout lane<16 -> M=r, lane>=16 -> M=8+r
  for (int i = 0; i < 2; i++) {
    int mt = m0 + wm * 32 + i * 16;
    for (int j = 0; j < 4; j++) {
      int n = n0 + wn * 64 + j * 16 + l16;
      for (int r = 0; r < 8; r++) {
        int m = mt + half * 8 + r;
        float v = acc[i][j][r];
        if (BIAS)  v += biasg[(long)b * biasBatch + m];
        if (RESID) v += residg[(long)b * rBatch + (long)m * ldd + n];
        D[(long)m * ldd + n] = v;
      }
    }
  }
}

// ---------------- in-place row softmax over last dim (row length Ts)
__global__ __launch_bounds__(256) void softmax_kernel(float* __restrict__ S, int Ts) {
  float* r = S + (long)blockIdx.x * Ts;
  __shared__ float sh[256];
  float mx = -3.0e38f;
  for (int i = threadIdx.x; i < Ts; i += 256) mx = fmaxf(mx, r[i]);
  sh[threadIdx.x] = mx; __syncthreads();
  for (int o = 128; o > 0; o >>= 1) {
    if (threadIdx.x < (unsigned)o) sh[threadIdx.x] = fmaxf(sh[threadIdx.x], sh[threadIdx.x + o]);
    __syncthreads();
  }
  mx = sh[0]; __syncthreads();
  float sum = 0.f;
  for (int i = threadIdx.x; i < Ts; i += 256) { float e = __expf(r[i] - mx); r[i] = e; sum += e; }
  sh[threadIdx.x] = sum; __syncthreads();
  for (int o = 128; o > 0; o >>= 1) {
    if (threadIdx.x < (unsigned)o) sh[threadIdx.x] += sh[threadIdx.x + o];
    __syncthreads();
  }
  float inv = 1.0f / sh[0];
  for (int i = threadIdx.x; i < Ts; i += 256) r[i] *= inv;
}

// ---------------------------------------------------------------------------
extern "C" void kernel_launch(void* const* d_in, const int* in_sizes, int n_in,
                              void* d_out, int out_size, void* d_ws, size_t ws_size,
                              hipStream_t stream) {
  (void)in_sizes; (void)n_in; (void)out_size; (void)ws_size;
  const int B = 4, C = 512, T = 2048;

  const float* x_c = (const float*)d_in[0];
  const float* x_s = (const float*)d_in[1];
  const float* Wn  = (const float*)d_in[2];
  const float* bn  = (const float*)d_in[3];
  const float* Wm  = (const float*)d_in[4];
  const float* bm  = (const float*)d_in[5];
  const float* Wl  = (const float*)d_in[6];
  const float* bl  = (const float*)d_in[7];
  const float* Wo  = (const float*)d_in[8];
  const float* bo  = (const float*)d_in[9];
  float* out = (float*)d_out;

  char* w = (char*)d_ws;
  auto carve = [&](size_t bytes) -> float* {
    char* p = w; w += (bytes + 255) & ~(size_t)255; return (float*)p;
  };
  float* alpha_c = carve((size_t)B * C * 4);
  float* beta_c  = carve((size_t)B * C * 4);
  float* alpha_s = carve((size_t)B * C * 4);
  float* beta_s  = carve((size_t)B * C * 4);
  float* Wn_a = carve((size_t)B * C * C * 4);
  float* bn_a = carve((size_t)B * C * 4);
  float* Wm_a = carve((size_t)B * C * C * 4);
  float* bm_a = carve((size_t)B * C * 4);
  float* Mn = carve((size_t)B * C * T * 4);
  float* Mm = carve((size_t)B * C * T * 4);
  float* Ml = carve((size_t)B * C * T * 4);
  float* Mo = carve((size_t)B * C * T * 4);
  float* S  = carve((size_t)B * T * T * 4);

  // 1) instance-norm statistics
  rowstats_kernel<<<B * C, 256, 0, stream>>>(x_c, alpha_c, beta_c, T);
  rowstats_kernel<<<B * C, 256, 0, stream>>>(x_s, alpha_s, beta_s, T);
  // 2) fold norm into conv weights (per batch)
  wadjust_kernel<<<dim3(C, B), 256, 0, stream>>>(Wn, bn, alpha_c, beta_c, Wn_a, bn_a, C);
  wadjust_kernel<<<dim3(C, B), 256, 0, stream>>>(Wm, bm, alpha_s, beta_s, Wm_a, bm_a, C);

  const dim3 blk(256);
  // 3) Mn = Wn' * x_c + bn' ; Mm = Wm' * x_s + bm' ; Ml = Wl * x_s + bl
  gemm_wmma_kernel<false, false, true, false><<<dim3(T / 128, C / 128, B), blk, 0, stream>>>(
      Wn_a, (long)C * C, x_c, (long)C * T, Mn, (long)C * T, bn_a, C, nullptr, 0,
      C, T, C, C, T, T);
  gemm_wmma_kernel<false, false, true, false><<<dim3(T / 128, C / 128, B), blk, 0, stream>>>(
      Wm_a, (long)C * C, x_s, (long)C * T, Mm, (long)C * T, bm_a, C, nullptr, 0,
      C, T, C, C, T, T);
  gemm_wmma_kernel<false, false, true, false><<<dim3(T / 128, C / 128, B), blk, 0, stream>>>(
      Wl, 0, x_s, (long)C * T, Ml, (long)C * T, bl, 0, nullptr, 0,
      C, T, C, C, T, T);
  // 4) S = Mn^T * Mm  (A stored [C][T] -> TRANSA), then softmax rows
  gemm_wmma_kernel<true, false, false, false><<<dim3(T / 128, T / 128, B), blk, 0, stream>>>(
      Mn, (long)C * T, Mm, (long)C * T, S, (long)T * T, nullptr, 0, nullptr, 0,
      T, T, C, T, T, T);
  softmax_kernel<<<B * T, 256, 0, stream>>>(S, T);
  // 5) Mo = Ml * A^T  (B stored [t][s] -> TRANSB)
  gemm_wmma_kernel<false, true, false, false><<<dim3(T / 128, C / 128, B), blk, 0, stream>>>(
      Ml, (long)C * T, S, (long)T * T, Mo, (long)C * T, nullptr, 0, nullptr, 0,
      C, T, T, T, T, T);
  // 6) out = Wo * Mo + bo + x_c
  gemm_wmma_kernel<false, false, true, true><<<dim3(T / 128, C / 128, B), blk, 0, stream>>>(
      Wo, 0, Mo, (long)C * T, out, (long)C * T, bo, 0, x_c, (long)C * T,
      C, T, C, C, T, T);
}
